// SurMoE_83468394431121
// MI455X (gfx1250) — matrix operand order repaired and checked by reading, not compiled, not run on hardware
//
#include <hip/hip_runtime.h>
#include <math.h>

// ---------------- problem constants ----------------
#define B_   32
#define N1_  331
#define N2_  256
#define D_   256
#define H_   8
#define DH_  32
#define S_   587          // N1 + N2
#define TD_  768          // 3*D
#define ST_  37           // ceil(S/16) key/query tiles
#define SPAD_ 608         // 19*32, padded key dim for K=32 chunks

typedef __attribute__((ext_vector_type(16))) _Float16 v16h;
typedef __attribute__((ext_vector_type(8)))  _Float16 v8h;
typedef __attribute__((ext_vector_type(8)))  float    v8f;

static __device__ __forceinline__ float elu_f(float x) {
  return x > 0.f ? x : expm1f(x);
}

// Build A operand (16xK-chunk, M = lane&15, K halves split by lane>=16) from
// a K-contiguous row pointer.  K = 2*aoff+e (aoff = 8 for lanes >= 16).
static __device__ __forceinline__ v16h load_a_row(const _Float16* rowp, int aoff) {
  v8h x = *(const v8h*)(rowp + 2 * aoff);
  v8h y = *(const v8h*)(rowp + 2 * aoff + 8);
  v16h a;
#pragma unroll
  for (int e = 0; e < 8; ++e) { a[e] = x[e]; a[8 + e] = y[e]; }
  return a;
}

// Build B operand (K-chunk x 16, N = lane&15, K = (lane>=16?16:0)+e) from a
// K-contiguous pointer already offset by the lane's K base.
static __device__ __forceinline__ v16h load_b_row(const _Float16* kp) {
  v8h x = *(const v8h*)(kp);
  v8h y = *(const v8h*)(kp + 8);
  v16h b;
#pragma unroll
  for (int e = 0; e < 8; ++e) { b[e] = x[e]; b[8 + e] = y[e]; }
  return b;
}

// ---------------- 1) fp32 -> fp16 staging ----------------
__global__ void convert_kernel(const float* __restrict__ x1, const float* __restrict__ x2,
                               const float* __restrict__ lin_w, const float* __restrict__ add_w2,
                               const float* __restrict__ add_w3, const float* __restrict__ attn_win,
                               const float* __restrict__ attn_wout,
                               _Float16* comb, _Float16* linh, _Float16* w2h,
                               _Float16* w3h, _Float16* winh, _Float16* wouth) {
  const long combN = (long)B_ * S_ * D_;
  const long n_lin = (long)D_ * D_;
  const long n_w3  = (long)N1_ * D_;
  const long n_win = (long)TD_ * D_;
  const long TOT = combN + n_lin + n_lin + n_w3 + n_win + n_lin;
  long i = (long)blockIdx.x * blockDim.x + threadIdx.x;
  long stride = (long)gridDim.x * blockDim.x;
  for (; i < TOT; i += stride) {
    long j = i;
    if (j < combN) {
      int d = (int)(j % D_);
      long t = j / D_;
      int r = (int)(t % S_);
      int b = (int)(t / S_);
      float v = (r < N1_) ? x1[((long)b * N1_ + r) * D_ + d]
                          : x2[((long)b * N2_ + (r - N1_)) * D_ + d];
      comb[j] = (_Float16)v;
      continue;
    }
    j -= combN;
    if (j < n_lin) { linh[j] = (_Float16)lin_w[j]; continue; }
    j -= n_lin;
    if (j < n_lin) { w2h[j] = (_Float16)add_w2[j]; continue; }
    j -= n_lin;
    if (j < n_w3)  { w3h[j] = (_Float16)add_w3[j]; continue; }
    j -= n_w3;
    if (j < n_win) { winh[j] = (_Float16)attn_win[j]; continue; }
    j -= n_win;
    wouth[j] = (_Float16)attn_wout[j];
  }
}

// ---------------- 2) router / gate: w (B,4) ----------------
__global__ __launch_bounds__(256) void gate_kernel(
    const float* __restrict__ x1, const float* __restrict__ x2,
    const float* __restrict__ r_w1, const float* __restrict__ r_b1,
    const float* __restrict__ r_g, const float* __restrict__ r_beta,
    const float* __restrict__ r_w2, const float* __restrict__ r_b2,
    float* __restrict__ wgate) {
  int b = blockIdx.x, t = threadIdx.x;     // 256 threads, t == feature index
  __shared__ float m[2 * D_];
  __shared__ float red[256];
  __shared__ float lg[4];

  { // means over rows
    float s = 0.f;
    const float* p = x1 + (long)b * N1_ * D_ + t;
    for (int n = 0; n < N1_; ++n) s += p[(long)n * D_];
    m[t] = s / (float)N1_;
    s = 0.f;
    const float* q = x2 + (long)b * N2_ * D_ + t;
    for (int n = 0; n < N2_; ++n) s += q[(long)n * D_];
    m[D_ + t] = s / (float)N2_;
  }
  __syncthreads();

  float hv = r_b1[t];
  const float* wr = r_w1 + (long)t * (2 * D_);
  for (int k = 0; k < 2 * D_; ++k) hv += m[k] * wr[k];

  red[t] = hv; __syncthreads();
  for (int o = 128; o > 0; o >>= 1) { if (t < o) red[t] += red[t + o]; __syncthreads(); }
  float mu = red[0] / (float)D_; __syncthreads();
  float dv = hv - mu;
  red[t] = dv * dv; __syncthreads();
  for (int o = 128; o > 0; o >>= 1) { if (t < o) red[t] += red[t + o]; __syncthreads(); }
  float var = red[0] / (float)D_; __syncthreads();

  float hn = dv * rsqrtf(var + 1e-5f) * r_g[t] + r_beta[t];
  float g = 0.5f * hn * (1.f + erff(hn * 0.70710678118654752f));  // exact GELU

#pragma unroll
  for (int e = 0; e < 4; ++e) {
    red[t] = g * r_w2[e * D_ + t]; __syncthreads();
    for (int o = 128; o > 0; o >>= 1) { if (t < o) red[t] += red[t + o]; __syncthreads(); }
    if (t == 0) lg[e] = red[0] + r_b2[e];
    __syncthreads();
  }
  if (t == 0) {
    float mx = fmaxf(fmaxf(lg[0], lg[1]), fmaxf(lg[2], lg[3]));
    float e0 = expf(lg[0] - mx), e1 = expf(lg[1] - mx), e2 = expf(lg[2] - mx), e3 = expf(lg[3] - mx);
    float s = e0 + e1 + e2 + e3;
    wgate[b * 4 + 0] = e0 / s; wgate[b * 4 + 1] = e1 / s;
    wgate[b * 4 + 2] = e2 / s; wgate[b * 4 + 3] = e3 / s;
  }
}

// ---------------- 3) generic C = act(A @ W^T + bias) via WMMA ----------------
// One wave computes a 16x64 output tile (4 accumulators): per K-chunk the A
// operand is loaded once and reused by 4 WMMAs.  A rows beyond `rows` are
// clamped (not masked): a garbage A-row m only affects D-row m, which is
// never stored — so no exec-mask dance in the inner loop.
// act: 0=none, 1=elu.
__global__ __launch_bounds__(32) void gemm_aw_kernel(
    const _Float16* __restrict__ A, int a_batch_row_stride, int rows, int lda,
    const _Float16* __restrict__ W, int ldw, int kTotal,
    const float* __restrict__ bias,
    float* __restrict__ outF, _Float16* __restrict__ outH,
    long out_batch_stride, int ldo, int act, int mtiles) {
  const int l = threadIdx.x;
  const int lanehi = l >> 4;
  const int ln = l & 15;
  const int m0 = (blockIdx.x % mtiles) * 16;
  const int n0 = (blockIdx.x / mtiles) * 64;       // 4 subtiles of 16
  const int batch = blockIdx.y;
  const int aoff = lanehi ? 8 : 0;

  int arow = m0 + ln; if (arow >= rows) arow = rows - 1;   // clamp, no mask
  const _Float16* Arow = A + ((long)batch * a_batch_row_stride + arow) * lda;
  const _Float16* Wr0 = W + (long)(n0 + ln) * ldw + (lanehi ? 16 : 0);

  v8f acc0 = {}, acc1 = {}, acc2 = {}, acc3 = {};
  for (int k0 = 0; k0 < kTotal; k0 += 32) {
    v16h a = load_a_row(Arow + k0, aoff);
    v16h b0 = load_b_row(Wr0 + k0);
    v16h b1 = load_b_row(Wr0 + (long)16 * ldw + k0);
    v16h b2 = load_b_row(Wr0 + (long)32 * ldw + k0);
    v16h b3 = load_b_row(Wr0 + (long)48 * ldw + k0);
    acc0 = __builtin_amdgcn_wmma_f32_16x16x32_f16(false, a, false, b0, (short)0, acc0, false, false);
    acc1 = __builtin_amdgcn_wmma_f32_16x16x32_f16(false, a, false, b1, (short)0, acc1, false, false);
    acc2 = __builtin_amdgcn_wmma_f32_16x16x32_f16(false, a, false, b2, (short)0, acc2, false, false);
    acc3 = __builtin_amdgcn_wmma_f32_16x16x32_f16(false, a, false, b3, (short)0, acc3, false, false);
  }

  v8f accs[4] = {acc0, acc1, acc2, acc3};
#pragma unroll
  for (int sub = 0; sub < 4; ++sub) {
    int n = n0 + sub * 16 + ln;
    float bi = bias[n];
#pragma unroll
    for (int r = 0; r < 8; ++r) {
      int mm = m0 + r + (lanehi ? 8 : 0);
      if (mm >= rows) continue;
      float v = accs[sub][r] + bi;
      if (act == 1) v = elu_f(v);
      long o = (long)batch * out_batch_stride + (long)mm * ldo + n;
      if (outF) outF[o] = v;
      if (outH) outH[o] = (_Float16)v;
    }
  }
}

// ---------------- 4) xout = x1l + elu(add_w3 @ x2 + add_b3[p]) ----------------
// out[b,p,d] tile: A = add_w3 (331x256), B[k][n] = comb[b, N1+k, n] (strided).
__global__ __launch_bounds__(32) void gemm_w3_kernel(
    const _Float16* __restrict__ w3h, const _Float16* __restrict__ comb,
    const float* __restrict__ add_b3, const float* __restrict__ t1,
    float* __restrict__ xout) {
  const int l = threadIdx.x;
  const int lanehi = l >> 4;
  const int ln = l & 15;
  const int m0 = (blockIdx.x % 21) * 16;
  const int n0 = (blockIdx.x / 21) * 16;
  const int b = blockIdx.y;
  const int aoff = lanehi ? 8 : 0;

  int p = m0 + ln; if (p >= N1_) p = N1_ - 1;              // clamp, no mask
  const _Float16* Arow = w3h + (long)p * D_;
  const int n = n0 + ln;
  const int kbase = lanehi ? 16 : 0;

  v8f acc = {};
  for (int k0 = 0; k0 < N2_; k0 += 32) {
    v16h a = load_a_row(Arow + k0, aoff);
    v16h bb;
#pragma unroll
    for (int e = 0; e < 16; ++e) {
      int kk = k0 + kbase + e;
      bb[e] = comb[((long)b * S_ + N1_ + kk) * D_ + n];
    }
    acc = __builtin_amdgcn_wmma_f32_16x16x32_f16(false, a, false, bb, (short)0, acc,
                                                 false, false);
  }

#pragma unroll
  for (int r = 0; r < 8; ++r) {
    int pm = m0 + r + (lanehi ? 8 : 0);
    if (pm >= N1_) continue;
    float x2l = elu_f(acc[r] + add_b3[pm]);   // row-indexed bias
    long o = ((long)b * N1_ + pm) * D_ + n;
    xout[o] = t1[o] + x2l;                    // t1 already = elu(x1@w2^T+b2)
  }
}

// ---------------- 5) attention per (b, h, 16-query tile) ----------------
__global__ __launch_bounds__(32) void attn_kernel(const _Float16* __restrict__ qkv,
                                                  _Float16* __restrict__ oh) {
  __shared__ __align__(16) float    sc[16 * SPAD_];
  __shared__ __align__(16) _Float16 pr[16 * SPAD_];
  const int l = threadIdx.x;
  const int lanehi = l >> 4;
  const int ln = l & 15;
  const int aoff = lanehi ? 8 : 0;
  const int qt = blockIdx.x, h = blockIdx.y, b = blockIdx.z;
  const int q0 = qt * 16;

  // A operand: q tile (K = DH = 32 exactly, one chunk).  Clamp OOB rows.
  int qrow = q0 + ln; if (qrow >= S_) qrow = S_ - 1;
  const _Float16* qp = qkv + ((long)b * S_ + qrow) * TD_ + h * DH_;
  v16h aq = load_a_row(qp, aoff);
  const float scale = 0.17677669529663687f;   // 1/sqrt(32)

  // scores: one WMMA per 16-key tile.  OOB keys are clamped, not masked:
  // their score columns (>= S_) are never read by the softmax below.
  for (int kt = 0; kt < ST_; ++kt) {
    int key = kt * 16 + ln; if (key >= S_) key = S_ - 1;
    const _Float16* kp = qkv + ((long)b * S_ + key) * TD_ + D_ + h * DH_ + (lanehi ? 16 : 0);
    v16h bk = load_b_row(kp);
    v8f s = {};
    s = __builtin_amdgcn_wmma_f32_16x16x32_f16(false, aq, false, bk, (short)0, s,
                                               false, false);
#pragma unroll
    for (int r = 0; r < 8; ++r)
      sc[(r + (lanehi ? 8 : 0)) * SPAD_ + kt * 16 + ln] = s[r] * scale;
  }
  __syncthreads();

  // row softmax (lane = query row)
  if (l < 16) {
    float* row = sc + l * SPAD_;
    float mx = -1e30f;
    for (int c = 0; c < S_; ++c) mx = fmaxf(mx, row[c]);
    float sum = 0.f;
    for (int c = 0; c < S_; ++c) { float e = expf(row[c] - mx); row[c] = e; sum += e; }
    float inv = 1.f / sum;
    _Float16* prow = pr + l * SPAD_;
    for (int c = 0; c < S_; ++c) prow[c] = (_Float16)(row[c] * inv);
    for (int c = S_; c < SPAD_; ++c) prow[c] = (_Float16)0.f;
  }
  __syncthreads();

  // o = attn @ v : K chunks of 32 keys, DH=32 -> two N halves
  v8f acc0 = {}, acc1 = {};
  for (int kc = 0; kc < SPAD_ / 32; ++kc) {
    int kb = kc * 32;
    v16h ap = load_a_row(pr + ln * SPAD_ + kb, aoff);
    int kkbase = kb + (lanehi ? 16 : 0);
    v16h bv0, bv1;
#pragma unroll
    for (int e = 0; e < 16; ++e) {
      int key = kkbase + e; if (key >= S_) key = S_ - 1;   // prob is 0 there
      const _Float16* vp = qkv + ((long)b * S_ + key) * TD_ + 2 * D_ + h * DH_;
      bv0[e] = vp[ln];
      bv1[e] = vp[16 + ln];
    }
    acc0 = __builtin_amdgcn_wmma_f32_16x16x32_f16(false, ap, false, bv0, (short)0, acc0,
                                                  false, false);
    acc1 = __builtin_amdgcn_wmma_f32_16x16x32_f16(false, ap, false, bv1, (short)0, acc1,
                                                  false, false);
  }
#pragma unroll
  for (int r = 0; r < 8; ++r) {
    int m = q0 + r + (lanehi ? 8 : 0);
    if (m >= S_) continue;
    long base = ((long)b * S_ + m) * D_ + h * DH_;
    oh[base + ln]      = (_Float16)acc0[r];
    oh[base + 16 + ln] = (_Float16)acc1[r];
  }
}

// ---------------- 6) final mixture combine ----------------
__global__ void combine_kernel(const float* __restrict__ wgate, const float* __restrict__ lo,
                               const float* __restrict__ xout, const float* __restrict__ o2,
                               const float* __restrict__ x1, const float* __restrict__ x2,
                               float* __restrict__ out) {
  long i = (long)blockIdx.x * blockDim.x + threadIdx.x;
  const long TOT = (long)B_ * S_ * D_;
  if (i >= TOT) return;
  int d = (int)(i % D_);
  long t = i / D_;
  int r = (int)(t % S_);
  int b = (int)(t / S_);
  float w0 = wgate[b * 4 + 0], w1 = wgate[b * 4 + 1];
  float w2 = wgate[b * 4 + 2], w3 = wgate[b * 4 + 3];
  float vlo = lo[i], vo2 = o2[i];
  if (r < N1_) {
    long ix = ((long)b * N1_ + r) * D_ + d;
    out[ix] = w0 * vlo + w1 * xout[ix] + w2 * vo2 + w3 * x1[ix];
  } else {
    int n2 = r - N1_;
    float vx = xout[((long)b * N1_ + n2) * D_ + d];
    float v3 = x2[((long)b * N2_ + n2) * D_ + d];
    out[(long)B_ * N1_ * D_ + ((long)b * N2_ + n2) * D_ + d] =
        w0 * vlo + w1 * vx + w2 * vo2 + w3 * v3;
  }
}

// ---------------- host orchestration ----------------
extern "C" void kernel_launch(void* const* d_in, const int* in_sizes, int n_in,
                              void* d_out, int out_size, void* d_ws, size_t ws_size,
                              hipStream_t stream) {
  const float* x1 = (const float*)d_in[0];
  const float* x2 = (const float*)d_in[1];
  const float* r_w1 = (const float*)d_in[2];
  const float* r_b1 = (const float*)d_in[3];
  const float* r_g = (const float*)d_in[4];
  const float* r_beta = (const float*)d_in[5];
  const float* r_w2 = (const float*)d_in[6];
  const float* r_b2 = (const float*)d_in[7];
  const float* lin_w = (const float*)d_in[8];
  const float* lin_b = (const float*)d_in[9];
  const float* add_w2 = (const float*)d_in[10];
  const float* add_b2 = (const float*)d_in[11];
  const float* add_w3 = (const float*)d_in[12];
  const float* add_b3 = (const float*)d_in[13];
  const float* attn_win = (const float*)d_in[14];
  const float* attn_bin = (const float*)d_in[15];
  const float* attn_wout = (const float*)d_in[16];
  const float* attn_bout = (const float*)d_in[17];
  float* out = (float*)d_out;

  // workspace carve-up (256B aligned)
  char* ws = (char*)d_ws;
  size_t off = 0;
  auto alloc = [&](size_t bytes) -> char* {
    char* p = ws + off;
    off = (off + bytes + 255) & ~(size_t)255;
    return p;
  };
  const long combN = (long)B_ * S_ * D_;
  float*     wgate = (float*)alloc(B_ * 4 * sizeof(float));
  _Float16*  comb  = (_Float16*)alloc(combN * sizeof(_Float16));
  _Float16*  qkvh  = (_Float16*)alloc((long)B_ * S_ * TD_ * sizeof(_Float16));
  float*     lo    = (float*)alloc(combN * sizeof(float));
  float*     t1    = (float*)alloc((long)B_ * N1_ * D_ * sizeof(float));
  float*     xout  = (float*)alloc((long)B_ * N1_ * D_ * sizeof(float));
  _Float16*  oh    = (_Float16*)alloc(combN * sizeof(_Float16));
  float*     o2    = (float*)alloc(combN * sizeof(float));
  _Float16*  linh  = (_Float16*)alloc((long)D_ * D_ * sizeof(_Float16));
  _Float16*  w2h   = (_Float16*)alloc((long)D_ * D_ * sizeof(_Float16));
  _Float16*  w3h   = (_Float16*)alloc((long)N1_ * D_ * sizeof(_Float16));
  _Float16*  winh  = (_Float16*)alloc((long)TD_ * D_ * sizeof(_Float16));
  _Float16*  wouth = (_Float16*)alloc((long)D_ * D_ * sizeof(_Float16));
  (void)ws_size; (void)in_sizes; (void)n_in; (void)out_size;

  // 1) f16 staging
  {
    long tot = combN + (long)D_ * D_ * 3 + (long)N1_ * D_ + (long)TD_ * D_;
    int blocks = (int)((tot + 255) / 256);
    convert_kernel<<<blocks, 256, 0, stream>>>(x1, x2, lin_w, add_w2, add_w3, attn_win,
                                               attn_wout, comb, linh, w2h, w3h, winh, wouth);
  }
  // 2) gate
  gate_kernel<<<B_, 256, 0, stream>>>(x1, x2, r_w1, r_b1, r_g, r_beta, r_w2, r_b2, wgate);

  const int Mtot = B_ * S_;            // 18784, divisible by 16
  const int mt_all = Mtot / 16;        // 1174
  // 3a) lo = elu(comb @ lin_w^T + lin_b)
  gemm_aw_kernel<<<dim3(mt_all * (D_ / 64)), 32, 0, stream>>>(
      comb, 0, Mtot, D_, linh, D_, D_, lin_b, lo, nullptr, 0, D_, 1, mt_all);
  // 3b) t1 = elu(x1 @ add_w2^T + add_b2)   (x1 = first N1 rows of each comb batch)
  gemm_aw_kernel<<<dim3(21 * (D_ / 64), B_), 32, 0, stream>>>(
      comb, S_, N1_, D_, w2h, D_, D_, add_b2, t1, nullptr, (long)N1_ * D_, D_, 1, 21);
  // 3c) qkv = comb @ attn_win^T + attn_bin  (f16 out)
  gemm_aw_kernel<<<dim3(mt_all * (TD_ / 64)), 32, 0, stream>>>(
      comb, 0, Mtot, D_, winh, D_, D_, attn_bin, nullptr, qkvh, 0, TD_, 0, mt_all);
  // 4) xout = t1 + elu(add_w3 @ x2 + add_b3)
  gemm_w3_kernel<<<dim3(21 * (D_ / 16), B_), 32, 0, stream>>>(w3h, comb, add_b3, t1, xout);
  // 5) attention -> oh (f16)
  attn_kernel<<<dim3(ST_, H_, B_), 32, 0, stream>>>(qkvh, oh);
  // 5b) o2 = oh @ attn_wout^T + attn_bout
  gemm_aw_kernel<<<dim3(mt_all * (D_ / 64)), 32, 0, stream>>>(
      oh, 0, Mtot, D_, wouth, D_, D_, attn_bout, o2, nullptr, 0, D_, 0, mt_all);
  // 6) mixture combine -> gene, img
  {
    int blocks = (int)((combN + 255) / 256);
    combine_kernel<<<blocks, 256, 0, stream>>>(wgate, lo, xout, o2, x1, x2, out);
  }
}
